// SimplifiedDGCNNWrapper_21208548508407
// MI455X (gfx1250) — matrix operand order repaired and checked by reading, not compiled, not run on hardware
//
#include <hip/hip_runtime.h>
#include <hip/hip_bf16.h>

// ---------------------------------------------------------------------------
// SimplifiedDGCNN forward for MI455X (gfx1250, wave32, WMMA).
// All GEMM-shaped math uses v_wmma_f32_16x16x32_f16 (f16 in, f32 accumulate).
// All K dimensions padded to multiples of 32 so fragment loads are plain
// aligned b128 vector loads (no predication).
// ---------------------------------------------------------------------------

typedef _Float16 half_t;
typedef __attribute__((ext_vector_type(16))) _Float16 v16h;
typedef __attribute__((ext_vector_type(8)))  _Float16 v8h;
typedef __attribute__((ext_vector_type(8)))  float    v8f;

#define NPTS      10000
#define KNN       20
#define NCLS      20
#define NQ        100
#define NQ_PAD    112      // queries padded to 7 WMMA row tiles
#define QDIM      256
#define DG_EPS    1e-5f

// ---------------- WMMA helpers (CDNA5 layouts) ------------------------------

__device__ __forceinline__ v8f wmma_f16(v16h a, v16h b, v8f c) {
  // D = A(16x32 f16) * B(32x16 f16) + C(16x16 f32)
  return __builtin_amdgcn_wmma_f32_16x16x32_f16(
      /*neg_a=*/false, a, /*neg_b=*/false, b,
      /*c_mod=*/(short)0, c, /*reuse_a=*/false, /*reuse_b=*/false);
}

// A fragment: 16x32 f16 tile from row-major [rows, stride] array (global or LDS).
// Lane l (<16): row = row0+l, K = kk+{0..15}
// Lane l (>=16): row = row0+l-16, K = kk+16+{0..15}
// stride and kk are multiples of 32 halfs -> the 32B load is aligned.
__device__ __forceinline__ v16h load_a_frag(const half_t* A, int stride,
                                            int row0, int kk, int lane) {
  const int row = row0 + (lane & 15);
  const int kb  = (lane < 16) ? 0 : 16;
  const half_t* p = A + (size_t)row * stride + kk + kb;
  return *(const v16h*)(p);
}

// B fragment: 32x16 f16 tile; B[k][col] = W[col][k], W row-major [cols, stride].
// Lane l: col = col0 + (l%16); K = kk+{0..15} (lanes<16) or kk+16+{0..15}.
// One aligned 32-byte load (backend splits into 2x b128).
__device__ __forceinline__ v16h load_b_frag(const half_t* W, int stride,
                                            int col0, int kk, int lane) {
  const int col   = col0 + (lane & 15);
  const int kbase = kk + ((lane < 16) ? 0 : 16);
  return *(const v16h*)(W + (size_t)col * stride + kbase);
}

// ---------------- small utility kernels -------------------------------------

__global__ void f32_to_f16_kernel(const float* __restrict__ src,
                                  half_t* __restrict__ dst, int n) {
  int t = blockIdx.x * blockDim.x + threadIdx.x;
  if (t < n) dst[t] = (half_t)src[t];
}

// dst [rows, Kpad] f16 <- src [rows, Kin] f32, zero pad K
__global__ void pad_w_f16_kernel(const float* __restrict__ src, int Kin,
                                 half_t* __restrict__ dst, int Kpad, int total) {
  int t = blockIdx.x * blockDim.x + threadIdx.x;
  if (t >= total) return;
  int r = t / Kpad, c = t - r * Kpad;
  dst[t] = (c < Kin) ? (half_t)src[r * Kin + c] : (half_t)0.f;
}

// dst[n*Cp + c] = (c < C) ? src[n*stride + c] : 0   (zero pad K to Cp)
__global__ void slice_to_f16_kernel(const float* __restrict__ src, int stride, int C,
                                    half_t* __restrict__ dst, int Cp) {
  int t = blockIdx.x * blockDim.x + threadIdx.x;
  if (t >= NPTS * Cp) return;
  int n = t / Cp, c = t - n * Cp;
  dst[t] = (c < C) ? (half_t)src[(size_t)n * stride + c] : (half_t)0.f;
}

// x0[n] = [coords.xyz, feats.w]; also f16 copy padded to 32 channels.
__global__ void build_x0_kernel(const float* __restrict__ coords,
                                const float* __restrict__ feats,
                                float* __restrict__ x0, half_t* __restrict__ x0h) {
  int n = blockIdx.x * blockDim.x + threadIdx.x;
  if (n >= NPTS) return;
  float v0 = coords[n * 3 + 0], v1 = coords[n * 3 + 1], v2 = coords[n * 3 + 2];
  float v3 = feats[n * 4 + 3];
  x0[n * 4 + 0] = v0; x0[n * 4 + 1] = v1; x0[n * 4 + 2] = v2; x0[n * 4 + 3] = v3;
  half_t* ph = x0h + n * 32;
  ph[0] = (half_t)v0; ph[1] = (half_t)v1; ph[2] = (half_t)v2; ph[3] = (half_t)v3;
#pragma unroll
  for (int c = 4; c < 32; ++c) ph[c] = (half_t)0.f;
}

__global__ void row_norms_kernel(const float* __restrict__ X, int stride, int C,
                                 float* __restrict__ norms) {
  int n = blockIdx.x * blockDim.x + threadIdx.x;
  if (n >= NPTS) return;
  const float* p = X + (size_t)n * stride;
  float s = 0.f;
  for (int c = 0; c < C; ++c) { float x = p[c]; s += x * x; }
  norms[n] = s;
}

__global__ void build_qh_kernel(const float* __restrict__ q, half_t* __restrict__ qh) {
  int t = blockIdx.x * blockDim.x + threadIdx.x;
  if (t >= NQ_PAD * QDIM) return;
  int row = t / QDIM, col = t - row * QDIM;
  qh[t] = (row < NQ) ? (half_t)q[row * QDIM + col] : (half_t)0.f;
}

// ---------------- KNN: WMMA Gram tiles + per-lane top-k ---------------------
// One wave per 16-query tile; NK = Cp/32 K-chunks is a template parameter so
// the hoisted query-tile A fragments are exactly sized (no spill, no guards).
// __launch_bounds__(32,1): single wave per block -> full VGPR budget.
// Outer candidate loop and the 16-candidate scan are kept rolled (the 20-deep
// insertion network below would otherwise be replicated per unrolled copy and
// blow up the hot loop past the per-WGP I$).
template <int NK>
__global__ __launch_bounds__(32, 1)
void knn_kernel(const half_t* __restrict__ Xh,
                const float* __restrict__ norms,
                int* __restrict__ idx) {
  constexpr int Cp = NK * 32;
  __shared__ __align__(32) float dtile[16 * 16];
  const int lane = threadIdx.x & 31;
  const int q0   = blockIdx.x * 16;

  v16h afr[NK];
#pragma unroll
  for (int kc = 0; kc < NK; ++kc)
    afr[kc] = load_a_frag(Xh, Cp, q0, kc * 32, lane);

  float kd[KNN]; int ki[KNN];
#pragma unroll
  for (int t = 0; t < KNN; ++t) { kd[t] = 3.0e38f; ki[t] = 0; }
  const float qn = norms[q0 + (lane & 15)];

#pragma clang loop unroll(disable)
  for (int c0 = 0; c0 < NPTS; c0 += 16) {
    v8f acc = {};
#pragma unroll
    for (int kc = 0; kc < NK; ++kc)
      acc = wmma_f16(afr[kc], load_b_frag(Xh, Cp, c0, kc * 32, lane), acc);
    __syncthreads();
    // D layout: col = lane%16, rows r + 8*(lane>=16). Fold |c|^2 - 2*inner.
    const int col = lane & 15;
    const float cn = norms[c0 + col];
#pragma unroll
    for (int r = 0; r < 8; ++r) {
      int rr = r + ((lane < 16) ? 0 : 8);
      dtile[rr * 16 + col] = cn - 2.0f * acc[r];
    }
    __syncthreads();
    if (lane < 16) {
#pragma clang loop unroll(disable)
      for (int c = 0; c < 16; ++c) {
        float dv = qn + dtile[lane * 16 + c];
        if (dv < kd[KNN - 1]) {      // threshold early-out: most candidates skip
          int iv = c0 + c;
#pragma unroll
          for (int t = 0; t < KNN; ++t) {
            if (dv < kd[t]) {
              float td = kd[t]; int ti = ki[t];
              kd[t] = dv; ki[t] = iv; dv = td; iv = ti;
            }
          }
        }
      }
    }
  }
  if (lane < 16) {
#pragma unroll
    for (int t = 0; t < KNN; ++t) idx[(q0 + lane) * KNN + t] = ki[t];
  }
}

// ---------------- EdgeConv MLP: implicit GEMM + max over neighbors ----------
// Block = 256 threads (8 waves) handles one point. 32 edge-feature rows
// (20 neighbors + 12 zero pad) built in LDS as f16. Each wave owns 16-wide
// Cout tiles; two WMMA row-tiles per Cout tile; BN + leaky-ReLU epilogue;
// max over neighbors via register max + __shfl_xor across lane halves.
// Wh is [Cout, KP] f16 with K zero-padded to KP (multiple of 32).
__global__ void edge_mlp_kernel(const float* __restrict__ X, int xstride, int Cin,
                                const int* __restrict__ idx,
                                const half_t* __restrict__ Wh,   // [Cout, KP] f16
                                const float* __restrict__ bn_g,
                                const float* __restrict__ bn_b,
                                const float* __restrict__ bn_m,
                                const float* __restrict__ bn_v,
                                float* __restrict__ out, int ostride, int Cout) {
  __shared__ __align__(32) half_t efs[32 * 512];   // 32 KB, max 2*Cin = 512
  const int n   = blockIdx.x;
  const int tid = threadIdx.x;
  const int K2  = 2 * Cin;
  const int KP  = (K2 < 32) ? 32 : K2;             // pad K to a WMMA chunk

  for (int e = tid; e < 32 * KP; e += blockDim.x) {
    int j = e / KP, c = e - j * KP;
    float val = 0.f;
    if (j < KNN && c < K2) {
      if (c < Cin) {
        int nb = idx[n * KNN + j];
        val = X[(size_t)nb * xstride + c] - X[(size_t)n * xstride + c];
      } else {
        val = X[(size_t)n * xstride + (c - Cin)];
      }
    }
    efs[e] = (half_t)val;
  }
  __syncthreads();

  const int lane = tid & 31, wave = tid >> 5;
  const int otiles = Cout >> 4;
  for (int t = wave; t < otiles; t += (blockDim.x >> 5)) {
    v8f acc1 = {};   // edge rows 0..15  (neighbors 0..15)
    v8f acc2 = {};   // edge rows 16..31 (neighbors 16..19 + pad)
    for (int kk = 0; kk < KP; kk += 32) {
      v16h bf = load_b_frag(Wh, KP, t * 16, kk, lane);
      v16h a1 = load_a_frag(efs, KP, 0,  kk, lane);
      v16h a2 = load_a_frag(efs, KP, 16, kk, lane);
      acc1 = wmma_f16(a1, bf, acc1);
      acc2 = wmma_f16(a2, bf, acc2);
    }
    const int col = t * 16 + (lane & 15);
    const float sc = bn_g[col] * rsqrtf(bn_v[col] + DG_EPS);
    const float sh = bn_b[col] - bn_m[col] * sc;
    float best = -3.0e38f;
#pragma unroll
    for (int r = 0; r < 8; ++r) {                  // rows 0..7 / 8..15: all valid
      float hv = acc1[r] * sc + sh;
      hv = (hv > 0.f) ? hv : 0.2f * hv;
      best = fmaxf(best, hv);
    }
    if (lane < 16) {                               // acc2 rows 0..3 = neighbors 16..19
#pragma unroll
      for (int r = 0; r < 4; ++r) {
        float hv = acc2[r] * sc + sh;
        hv = (hv > 0.f) ? hv : 0.2f * hv;
        best = fmaxf(best, hv);
      }
    }
    best = fmaxf(best, __shfl_xor(best, 16, 32));  // merge lane halves (rows 0-7 vs 8-15)
    if (lane < 16) out[(size_t)n * ostride + col] = best;
  }
}

// ---------------- Generic tiled WMMA GEMM with BN/activation epilogue -------
// C[m, o] = act(BN(sum_k A[m,k] * W[o,k])). A row-block (16 x K) staged in LDS
// and shared by 8 waves; wave w handles Cout tile t = blockIdx.y*8 + w.
__global__ void gemm16_kernel(const half_t* __restrict__ A,   // [M, K] f16
                              const half_t* __restrict__ W,   // [O, K] f16
                              int K, int O, int Otiles, int Mvalid,
                              const float* __restrict__ g, const float* __restrict__ b,
                              const float* __restrict__ m, const float* __restrict__ v,
                              int lrelu,
                              float* __restrict__ outF, half_t* __restrict__ outH) {
  __shared__ __align__(32) half_t As[16 * 1024];   // 32 KB, K <= 1024
  const int m0 = blockIdx.x * 16;
  {
    // vectorized staging: 16 rows x K halfs as 16-byte chunks
    const uint4* src = (const uint4*)(A + (size_t)m0 * K);
    uint4* dst = (uint4*)As;
    const int nchunks = (16 * K) >> 3;             // K % 8 == 0 always
    for (int e = threadIdx.x; e < nchunks; e += blockDim.x) dst[e] = src[e];
  }
  __syncthreads();

  const int lane = threadIdx.x & 31, wave = threadIdx.x >> 5;
  const int t = blockIdx.y * 8 + wave;
  if (t >= Otiles) return;

  v8f acc = {};
  for (int kk = 0; kk < K; kk += 32) {
    __builtin_prefetch(W + (size_t)(t * 16 + (lane & 15)) * K + kk + 256, 0, 1);
    v16h a  = load_a_frag(As, K, 0, kk, lane);
    v16h bf = load_b_frag(W, K, t * 16, kk, lane);
    acc = wmma_f16(a, bf, acc);
  }

  const int col = t * 16 + (lane & 15);
  float sc = 1.f, sh = 0.f;
  if (g) {
    sc = g[col] * rsqrtf(v[col] + DG_EPS);
    sh = b[col] - m[col] * sc;
  }
#pragma unroll
  for (int r = 0; r < 8; ++r) {
    int row = m0 + r + ((lane < 16) ? 0 : 8);
    float val = acc[r] * sc + sh;
    if (lrelu) val = (val > 0.f) ? val : 0.2f * val;
    if (row < Mvalid) {
      if (outF) outF[(size_t)row * O + col] = val;
      if (outH) outH[(size_t)row * O + col] = (half_t)val;
    }
  }
}

// ---------------- tiny tail kernels (VALU is fine here) ---------------------

__global__ void sem_kernel(const float* __restrict__ feat,
                           const float* __restrict__ Wsem,
                           const float* __restrict__ bsem,
                           float* __restrict__ out) {
  int t = blockIdx.x * blockDim.x + threadIdx.x;
  if (t >= NPTS * NCLS) return;
  int n = t / NCLS, c = t - n * NCLS;
  const float* f = feat + (size_t)n * 256;
  const float* w = Wsem + (size_t)c * 256;
  float s = 0.f;
  for (int k = 0; k < 256; ++k) s += f[k] * w[k];
  out[t] = s + bsem[c];
}

__global__ void logits_kernel(const float* __restrict__ q,
                              const float* __restrict__ clsW,
                              const float* __restrict__ clsb,
                              float* __restrict__ out) {
  int t = blockIdx.x * blockDim.x + threadIdx.x;
  if (t >= NQ * (NCLS + 1)) return;
  int qi = t / (NCLS + 1), c = t - qi * (NCLS + 1);
  const float* qp = q + (size_t)qi * QDIM;
  const float* wp = clsW + (size_t)c * QDIM;
  float s = 0.f;
  for (int k = 0; k < QDIM; ++k) s += qp[k] * wp[k];
  out[t] = s + clsb[c];
}

// ---------------------------------------------------------------------------

static inline size_t align256(size_t x) { return (x + 255) & ~(size_t)255; }

extern "C" void kernel_launch(void* const* d_in, const int* in_sizes, int n_in,
                              void* d_out, int out_size, void* d_ws, size_t ws_size,
                              hipStream_t stream) {
  (void)in_sizes; (void)n_in; (void)out_size; (void)ws_size;

  // Sorted-key pytree flatten order of setup_inputs():
  // 0 coords, 1 feats,
  // 2..6   conv5: W,b,g,m,v
  // 7..9   dec:   cls_W, cls_b, q
  // 10..14 ec1:   W,b,g,m,v
  // 15..19 ec2,  20..24 ec3,  25..29 ec4 (same order)
  // 30..34 feat:  W,b,g,m,v
  // 35..36 sem:   W,b
  const float* coords = (const float*)d_in[0];
  const float* feats  = (const float*)d_in[1];
  const float* c5W = (const float*)d_in[2];
  const float* c5b = (const float*)d_in[3], *c5g = (const float*)d_in[4];
  const float* c5m = (const float*)d_in[5], *c5v = (const float*)d_in[6];
  const float* clsW = (const float*)d_in[7];
  const float* clsb = (const float*)d_in[8];
  const float* qW   = (const float*)d_in[9];
  const float* ecW[4]  = { (const float*)d_in[10], (const float*)d_in[15],
                           (const float*)d_in[20], (const float*)d_in[25] };
  const float* ecb[4]  = { (const float*)d_in[11], (const float*)d_in[16],
                           (const float*)d_in[21], (const float*)d_in[26] };
  const float* ecg[4]  = { (const float*)d_in[12], (const float*)d_in[17],
                           (const float*)d_in[22], (const float*)d_in[27] };
  const float* ecm[4]  = { (const float*)d_in[13], (const float*)d_in[18],
                           (const float*)d_in[23], (const float*)d_in[28] };
  const float* ecv[4]  = { (const float*)d_in[14], (const float*)d_in[19],
                           (const float*)d_in[24], (const float*)d_in[29] };
  const float* ftW = (const float*)d_in[30];
  const float* ftb = (const float*)d_in[31], *ftg = (const float*)d_in[32];
  const float* ftm = (const float*)d_in[33], *ftv = (const float*)d_in[34];
  const float* semW = (const float*)d_in[35];
  const float* semb = (const float*)d_in[36];

  // output: pred_logits [100,21] | pred_masks [100,10000] | sem_logits [10000,20]
  float* outLogits = (float*)d_out;
  float* outMasks  = outLogits + NQ * (NCLS + 1);
  float* outSem    = outMasks + (size_t)NQ * NPTS;

  // ---- workspace carve ----
  char* wsb = (char*)d_ws;
  size_t off = 0;
  auto carve = [&](size_t bytes) -> void* {
    void* p = wsb + off; off = align256(off + bytes); return p;
  };
  float*  xc    = (float*) carve((size_t)NPTS * 512 * 4);   // [x1|x2|x3|x4] channel concat
  float*  x0    = (float*) carve((size_t)NPTS * 4 * 4);
  half_t* x0h   = (half_t*)carve((size_t)NPTS * 32 * 2);    // padded to Cp=32
  half_t* xh    = (half_t*)carve((size_t)NPTS * 128 * 2);   // reused per layer (Cp<=128)
  half_t* xch   = (half_t*)carve((size_t)NPTS * 512 * 2);
  half_t* x5h   = (half_t*)carve((size_t)NPTS * 1024 * 2);
  float*  featF = (float*) carve((size_t)NPTS * 256 * 4);
  half_t* featH = (half_t*)carve((size_t)NPTS * 256 * 2);
  int*    nidx  = (int*)   carve((size_t)NPTS * KNN * 4);
  float*  norms = (float*) carve((size_t)NPTS * 4);
  half_t* W1h = (half_t*)carve((size_t)64   * 32  * 2);     // K padded 8 -> 32
  half_t* W2h = (half_t*)carve((size_t)64   * 128 * 2);
  half_t* W3h = (half_t*)carve((size_t)128  * 128 * 2);
  half_t* W4h = (half_t*)carve((size_t)256  * 256 * 2);
  half_t* W5h = (half_t*)carve((size_t)1024 * 512 * 2);
  half_t* Wfh = (half_t*)carve((size_t)256  * 1024 * 2);
  half_t* qh  = (half_t*)carve((size_t)NQ_PAD * QDIM * 2);
  half_t* ecWh[4] = { W1h, W2h, W3h, W4h };

  const int T = 256;
  auto nb = [](int n, int t) { return (n + t - 1) / t; };

  // ---- weight / query f16 conversion (ec1.W padded 8 -> 32 in K) ----
  pad_w_f16_kernel<<<nb(64 * 32, T), T, 0, stream>>>(ecW[0], 8, W1h, 32, 64 * 32);
  f32_to_f16_kernel<<<nb(64 * 128, T),   T, 0, stream>>>(ecW[1], W2h, 64 * 128);
  f32_to_f16_kernel<<<nb(128 * 128, T),  T, 0, stream>>>(ecW[2], W3h, 128 * 128);
  f32_to_f16_kernel<<<nb(256 * 256, T),  T, 0, stream>>>(ecW[3], W4h, 256 * 256);
  f32_to_f16_kernel<<<nb(1024 * 512, T), T, 0, stream>>>(c5W, W5h, 1024 * 512);
  f32_to_f16_kernel<<<nb(256 * 1024, T), T, 0, stream>>>(ftW, Wfh, 256 * 1024);
  build_qh_kernel<<<nb(NQ_PAD * QDIM, T), T, 0, stream>>>(qW, qh);

  // ---- x0 = [coords, feats[:,3]] ----
  build_x0_kernel<<<nb(NPTS, T), T, 0, stream>>>(coords, feats, x0, x0h);

  // ---- EdgeConv stack; outputs land in channel slices of xc ----
  const int   Cin[4]   = { 4, 64, 64, 128 };
  const int   Cp[4]    = { 32, 64, 64, 128 };
  const int   Cout[4]  = { 64, 64, 128, 256 };
  const int   chOff[4] = { 0, 64, 128, 256 };
  for (int L = 0; L < 4; ++L) {
    const float*  Xf = (L == 0) ? x0  : (xc + chOff[L - 1]);
    const int     xs = (L == 0) ? 4   : 512;
    const half_t* Xh = (L == 0) ? x0h : xh;
    if (L > 0) {
      slice_to_f16_kernel<<<nb(NPTS * Cp[L], T), T, 0, stream>>>(Xf, xs, Cin[L], xh, Cp[L]);
    }
    row_norms_kernel<<<nb(NPTS, T), T, 0, stream>>>(Xf, xs, Cin[L], norms);
    if (Cp[L] == 32) {
      knn_kernel<1><<<NPTS / 16, 32, 0, stream>>>(Xh, norms, nidx);
    } else if (Cp[L] == 64) {
      knn_kernel<2><<<NPTS / 16, 32, 0, stream>>>(Xh, norms, nidx);
    } else {
      knn_kernel<4><<<NPTS / 16, 32, 0, stream>>>(Xh, norms, nidx);
    }
    edge_mlp_kernel<<<NPTS, T, 0, stream>>>(Xf, xs, Cin[L], nidx, ecWh[L],
                                            ecg[L], ecb[L], ecm[L], ecv[L],
                                            xc + chOff[L], 512, Cout[L]);
  }

  // ---- conv5: [N,512] x [512->1024], BN + lrelu -> x5h (f16 only) ----
  f32_to_f16_kernel<<<nb(NPTS * 512, T), T, 0, stream>>>(xc, xch, NPTS * 512);
  gemm16_kernel<<<dim3(NPTS / 16, 8), T, 0, stream>>>(
      xch, W5h, 512, 1024, 64, NPTS, c5g, c5b, c5m, c5v, 1, nullptr, x5h);

  // ---- feat: [N,1024] x [1024->256], BN -> featF (f32) + featH (f16) ----
  gemm16_kernel<<<dim3(NPTS / 16, 2), T, 0, stream>>>(
      x5h, Wfh, 1024, 256, 16, NPTS, ftg, ftb, ftm, ftv, 0, featF, featH);

  // ---- sem logits and pred logits ----
  sem_kernel<<<nb(NPTS * NCLS, T), T, 0, stream>>>(featF, semW, semb, outSem);
  logits_kernel<<<nb(NQ * (NCLS + 1), T), T, 0, stream>>>(qW, clsW, clsb, outLogits);

  // ---- pred masks: q [100(pad112),256] x feat^T [256,10000] via WMMA ----
  gemm16_kernel<<<dim3(NQ_PAD / 16, 79), T, 0, stream>>>(
      qh, featH, 256, NPTS, NPTS / 16, NQ,
      nullptr, nullptr, nullptr, nullptr, 0, outMasks, nullptr);
}
